// Fp8DotGeneralOp_58428735095555
// MI455X (gfx1250) — 8-bit matrix operand order repaired and checked by reading, compiled, not run on hardware
//
#include <hip/hip_runtime.h>

typedef __attribute__((ext_vector_type(16))) int   v16i;
typedef __attribute__((ext_vector_type(8)))  float v8f;

static constexpr int M = 16384;
static constexpr int N = 4096;
static constexpr int K = 1024;
static constexpr float FP8_E4M3_MAX = 448.0f;

#define BM 128
#define BN 128
#define BK 128
#define LDS_STRIDE 144

#if __has_builtin(__builtin_amdgcn_global_load_async_to_lds_b128) && \
    __has_builtin(__builtin_amdgcn_s_wait_asynccnt)
#define USE_ASYNC 1
#else
#define USE_ASYNC 0
#endif

#define AS_GLOBAL __attribute__((address_space(1)))
#define AS_LOCAL  __attribute__((address_space(3)))

typedef int v4i_raw __attribute__((vector_size(16)));
typedef AS_GLOBAL v4i_raw* g_v4i_p;
typedef AS_LOCAL  v4i_raw* l_v4i_p;

__device__ __forceinline__ float exp2i(int e) {
  return __uint_as_float((unsigned)(e + 127) << 23);
}

__device__ __forceinline__ unsigned f32_to_e4m3(float x) {
  unsigned xb  = __float_as_uint(x);
  unsigned sgn = (xb >> 24) & 0x80u;
  float ax = __uint_as_float(xb & 0x7FFFFFFFu);
  if (ax > FP8_E4M3_MAX) ax = FP8_E4M3_MAX;
  if (!(ax >= 0x1p-10f)) return sgn;
  unsigned b = __float_as_uint(ax);
  int ebits = (int)(b >> 23) - 127;
  int eq = ebits < -6 ? -6 : ebits;
  long long mant = (long long)((b & 0x7FFFFFu) | 0x800000u);
  int shift = (eq - 3) - (ebits - 23);
  long long rem  = mant & ((1LL << shift) - 1);
  long long half = 1LL << (shift - 1);
  long long q = mant >> shift;
  if (rem > half || (rem == half && (q & 1))) q++;
  float val = (float)q * exp2i(eq - 3);
  if (val > FP8_E4M3_MAX) val = FP8_E4M3_MAX;
  unsigned vb = __float_as_uint(val);
  int ve = (int)(vb >> 23) - 127;
  unsigned enc;
  if (val == 0.0f)  enc = 0u;
  else if (ve < -6) enc = (unsigned)(val * 512.0f);
  else              enc = ((unsigned)(ve + 7) << 3) | ((vb >> 20) & 7u);
  return sgn | enc;
}

__global__ void k_init(float* wsf) {
  if (threadIdx.x < 8) wsf[threadIdx.x] = 0.0f;
}

__global__ void k_amax(const float* __restrict__ x, size_t n, float* __restrict__ out_amax) {
  __shared__ float red[256];
  float m = 0.0f;
  for (size_t i = (size_t)blockIdx.x * blockDim.x + threadIdx.x; i < n;
       i += (size_t)gridDim.x * blockDim.x)
    m = fmaxf(m, fabsf(x[i]));
  red[threadIdx.x] = m;
  __syncthreads();
  for (int s = 128; s > 0; s >>= 1) {
    if ((int)threadIdx.x < s)
      red[threadIdx.x] = fmaxf(red[threadIdx.x], red[threadIdx.x + s]);
    __syncthreads();
  }
  if (threadIdx.x == 0)
    atomicMax((unsigned int*)out_amax, __float_as_uint(red[0]));
}

__device__ __forceinline__ float compute_scale_dev(float amax, float old_scale) {
  float e  = floorf(log2f(FP8_E4M3_MAX / amax));
  float sf = roundf(exp2f(fabsf(e)));
  if (!(amax > 0.0f))            sf = old_scale;
  if (!__builtin_isfinite(amax)) sf = old_scale;
  if (e < 0.0f) sf = 1.0f / sf;
  return 1.0f / sf;
}

__global__ void k_scales(const float* __restrict__ in_scale, const float* __restrict__ kn_scale,
                         float* __restrict__ wsf) {
  if (threadIdx.x == 0 && blockIdx.x == 0) {
    float sa = compute_scale_dev(wsf[0], in_scale[0]);
    float sb = compute_scale_dev(wsf[1], kn_scale[0]);
    wsf[2] = sa;  wsf[3] = sb;
    wsf[4] = 1.0f / sa;  wsf[5] = 1.0f / sb;
    wsf[6] = sa * sb;
  }
}

__global__ void k_quant_rowmajor(const float* __restrict__ x, unsigned char* __restrict__ q,
                                 const float* __restrict__ wsf) {
  const float qmul = wsf[4];
  size_t base = ((size_t)blockIdx.x * blockDim.x + threadIdx.x) * 16;
  const float4* xv = (const float4*)(x + base);
  unsigned w[4];
#pragma unroll
  for (int v = 0; v < 4; v++) {
    float4 f = xv[v];
    unsigned b0 = f32_to_e4m3(f.x * qmul);
    unsigned b1 = f32_to_e4m3(f.y * qmul);
    unsigned b2 = f32_to_e4m3(f.z * qmul);
    unsigned b3 = f32_to_e4m3(f.w * qmul);
    w[v] = b0 | (b1 << 8) | (b2 << 16) | (b3 << 24);
  }
  *(uint4*)(q + base) = make_uint4(w[0], w[1], w[2], w[3]);
}

__global__ void k_quant_transpose(const float* __restrict__ x, unsigned char* __restrict__ q,
                                  const float* __restrict__ wsf) {
  const float qmul = wsf[5];
  size_t id = (size_t)blockIdx.x * blockDim.x + threadIdx.x;
  int n  = (int)(id >> 6);
  int k0 = ((int)id & 63) << 4;
  unsigned w[4];
#pragma unroll
  for (int v = 0; v < 4; v++) {
    unsigned acc = 0;
#pragma unroll
    for (int j = 0; j < 4; j++) {
      float f = x[(size_t)(k0 + v * 4 + j) * N + n];
      acc |= f32_to_e4m3(f * qmul) << (8 * j);
    }
    w[v] = acc;
  }
  *(uint4*)(q + (size_t)n * K + k0) = make_uint4(w[0], w[1], w[2], w[3]);
}

__global__ __launch_bounds__(256)
void k_gemm_fp8(const unsigned char* __restrict__ Aq,
                const unsigned char* __restrict__ Bq,
                float* __restrict__ C,
                const float* __restrict__ wsf) {
#if USE_ASYNC
  __shared__ unsigned char lA[2][BM * LDS_STRIDE];
  __shared__ unsigned char lB[2][BN * LDS_STRIDE];
#else
  __shared__ unsigned char lA[1][BM * LDS_STRIDE];
  __shared__ unsigned char lB[1][BN * LDS_STRIDE];
#endif

  const int tid  = threadIdx.x;
  const int lane = tid & 31;
  const int wave = tid >> 5;
  const int wm   = (wave >> 1) * 32;
  const int wn   = (wave & 1) * 64;
  const int h    = lane >> 4;
  const int l16  = lane & 15;

  const int bm = blockIdx.x * BM;
  const int bn = blockIdx.y * BN;
  const float outscale = wsf[6];

  const int lrow = tid >> 1;
  const int lcol = (tid & 1) * 64;
  const unsigned char* gAbase = Aq + (size_t)(bm + lrow) * K + lcol;
  const unsigned char* gBbase = Bq + (size_t)(bn + lrow) * K + lcol;
  const int sOff = lrow * LDS_STRIDE + lcol;

  v8f zero = {};
  v8f acc[2][4];
#pragma unroll
  for (int i = 0; i < 2; i++)
#pragma unroll
    for (int j = 0; j < 4; j++) acc[i][j] = zero;

#if USE_ASYNC
  auto stage = [&](int kk, int b) {
#pragma unroll
    for (int i = 0; i < 4; ++i) {
      __builtin_amdgcn_global_load_async_to_lds_b128(
          (g_v4i_p)(v4i_raw*)(gAbase + kk + i * 16),
          (l_v4i_p)(v4i_raw*)(&lA[b][sOff + i * 16]), 0, 0);
      __builtin_amdgcn_global_load_async_to_lds_b128(
          (g_v4i_p)(v4i_raw*)(gBbase + kk + i * 16),
          (l_v4i_p)(v4i_raw*)(&lB[b][sOff + i * 16]), 0, 0);
    }
  };
  stage(0, 0);
  __builtin_amdgcn_s_wait_asynccnt(0);
  __syncthreads();
#endif

  for (int step = 0; step < K / BK; ++step) {
    const int kk = step * BK;
#if USE_ASYNC
    const int buf = step & 1;
    if (step + 1 < K / BK) stage(kk + BK, buf ^ 1);
#else
    const int buf = 0;
    const unsigned char* gA = gAbase + kk;
    const unsigned char* gB = gBbase + kk;
    uint4 a0 = ((const uint4*)gA)[0];
    uint4 a1 = ((const uint4*)gA)[1];
    uint4 a2 = ((const uint4*)gA)[2];
    uint4 a3 = ((const uint4*)gA)[3];
    uint4 b0 = ((const uint4*)gB)[0];
    uint4 b1 = ((const uint4*)gB)[1];
    uint4 b2 = ((const uint4*)gB)[2];
    uint4 b3 = ((const uint4*)gB)[3];
    __syncthreads();
    uint4* sA = (uint4*)&lA[0][sOff];
    uint4* sB = (uint4*)&lB[0][sOff];
    sA[0] = a0; sA[1] = a1; sA[2] = a2; sA[3] = a3;
    sB[0] = b0; sB[1] = b1; sB[2] = b2; sB[3] = b3;
    __syncthreads();
#endif

    v16i af[2], bf[4];
#pragma unroll
    for (int i = 0; i < 2; i++) {
      const unsigned char* p = &lA[buf][(wm + i * 16 + l16) * LDS_STRIDE + h * 16];
#pragma unroll
      for (int c = 0; c < 8; c++) {
        uint2 d = *(const uint2*)(p + (c >> 1) * 32 + (c & 1) * 8);
        af[i][2 * c]     = (int)d.x;
        af[i][2 * c + 1] = (int)d.y;
      }
    }
#pragma unroll
    for (int j = 0; j < 4; j++) {
      const unsigned char* p = &lB[buf][(wn + j * 16 + l16) * LDS_STRIDE + h * 16];
#pragma unroll
      for (int g = 0; g < 4; g++) {
        uint4 d = *(const uint4*)(p + g * 32);
        bf[j][4 * g]     = (int)d.x;
        bf[j][4 * g + 1] = (int)d.y;
        bf[j][4 * g + 2] = (int)d.z;
        bf[j][4 * g + 3] = (int)d.w;
      }
    }
#pragma unroll
    for (int i = 0; i < 2; i++)
#pragma unroll
      for (int j = 0; j < 4; j++)
        acc[i][j] = __builtin_amdgcn_wmma_f32_16x16x128_fp8_fp8(
            af[i], bf[j], (short)0, acc[i][j], false, false);

#if USE_ASYNC
    __builtin_amdgcn_s_wait_asynccnt(0);
    __syncthreads();
#endif
  }

#pragma unroll
  for (int i = 0; i < 2; i++) {
#pragma unroll
    for (int j = 0; j < 4; j++) {
      const int row0 = bm + wm + i * 16 + h * 8;
      const int col  = bn + wn + j * 16 + l16;
#pragma unroll
      for (int r = 0; r < 8; r++)
        C[(size_t)(row0 + r) * N + col] = acc[i][j][r] * outscale;
    }
  }
}

extern "C" void kernel_launch(void* const* d_in, const int* in_sizes, int n_in,
                              void* d_out, int out_size, void* d_ws, size_t ws_size,
                              hipStream_t stream) {
  const float* lhs      = (const float*)d_in[0];
  const float* rhs      = (const float*)d_in[1];
  const float* in_scale = (const float*)d_in[2];
  const float* kn_scale = (const float*)d_in[3];
  float* out = (float*)d_out;

  float* wsf = (float*)d_ws;
  unsigned char* qA = (unsigned char*)d_ws + 256;
  unsigned char* qB = qA + (size_t)M * K;

  k_init<<<1, 32, 0, stream>>>(wsf);
  k_amax<<<1024, 256, 0, stream>>>(lhs, (size_t)M * K, wsf + 0);
  k_amax<<<512, 256, 0, stream>>>(rhs, (size_t)K * N, wsf + 1);
  k_scales<<<1, 1, 0, stream>>>(in_scale, kn_scale, wsf);
  k_quant_rowmajor<<<(M * K / 16) / 256, 256, 0, stream>>>(lhs, qA, wsf);
  k_quant_transpose<<<(N * K / 16) / 256, 256, 0, stream>>>(rhs, qB, wsf);
  dim3 grid(M / BM, N / BN);
  k_gemm_fp8<<<grid, 256, 0, stream>>>(qA, qB, out, wsf);
}
